// GAT_32384053412574
// MI455X (gfx1250) — matrix operand order repaired and checked by reading, not compiled, not run on hardware
//
#include <hip/hip_runtime.h>

typedef __attribute__((ext_vector_type(16))) _Float16 v16h;
typedef __attribute__((ext_vector_type(8)))  float    v8f;
typedef unsigned int u32x4 __attribute__((ext_vector_type(4)));
typedef int          i32x8 __attribute__((ext_vector_type(8)));
typedef int          i32x4 __attribute__((ext_vector_type(4)));

#define N_NODES 4096
#define F_IN    512
#define HEADS   8
#define D_HEAD  8
#define F_HID   64     // HEADS * D_HEAD
#define NCLASS  16
#define LEAKY   0.2f

#if defined(__has_builtin)
#if __has_builtin(__builtin_amdgcn_tensor_load_to_lds) && __has_builtin(__builtin_amdgcn_s_wait_tensorcnt)
#define HAS_TDM 1
#endif
#endif
#ifndef HAS_TDM
#define HAS_TDM 0
#endif

// ---------------- workspace layout (bytes) ----------------
static constexpr size_t OFF_XH    = 0;                      // f16 [4096,512]
static constexpr size_t OFF_W1H   = OFF_XH    + 4194304;    // f16 [512,64]
static constexpr size_t OFF_H1    = OFF_W1H   + 65536;      // f32 [4096,64]
static constexpr size_t OFF_H1H   = OFF_H1    + 1048576;    // f16 [4096,64]
static constexpr size_t OFF_ES1   = OFF_H1H   + 524288;     // f32 [8,4096]
static constexpr size_t OFF_ED1   = OFF_ES1   + 131072;     // f32 [8,4096]
static constexpr size_t OFF_MR1   = OFF_ED1   + 131072;     // f32 [8,4096]
static constexpr size_t OFF_OUT1  = OFF_MR1   + 131072;     // f32 [4096,64]
static constexpr size_t OFF_EH    = OFF_OUT1  + 1048576;    // f16 [4096,64]
static constexpr size_t OFF_W2H   = OFF_EH    + 524288;     // f16 [64,16]
static constexpr size_t OFF_H2    = OFF_W2H   + 2048;       // f32 [4096,16]
static constexpr size_t OFF_H2H   = OFF_H2    + 262144;     // f16 [4096,16]
static constexpr size_t OFF_ES2   = OFF_H2H   + 131072;     // f32 [4096]
static constexpr size_t OFF_ED2   = OFF_ES2   + 16384;      // f32 [4096]
static constexpr size_t OFF_MR2   = OFF_ED2   + 16384;      // f32 [4096]

// ---------------- TDM tile loader (CDNA5 Tensor Data Mover) ----------------
// Loads a [tile_rows x tile_cols] f32 tile (row stride = stride_elems) from
// global memory into LDS, inserting 4 dwords of padding after every row
// (pad_interval must equal tile_cols). D# per CDNA5 ISA.
// This toolchain (clang-23 / therock-10.0) uses the 6-arg builtin form:
// (uint32x4 g0, int32x8 g1, int32x4 g2, int32x4 g3, int32x8 g4, i32 cpol).
#if HAS_TDM
__device__ __forceinline__ void tdm_load_tile(const float* gsrc, void* lds_ptr,
                                              int tile_cols, int tile_rows,
                                              int stride_elems, int pad_code) {
    unsigned long long ga = (unsigned long long)(size_t)gsrc;
    u32x4 g0;
    g0[0] = 1u;                                            // count=1 (valid), no gather
    g0[1] = (unsigned)(size_t)lds_ptr;                     // lds_addr (bytes)
    g0[2] = (unsigned)(ga & 0xffffffffull);                // global_addr[31:0]
    g0[3] = ((unsigned)(ga >> 32) & 0x01ffffffu)           // global_addr[56:32]
          | 0x80000000u;                                   // type = 2 ("image")
    int w0 = (2 << 16)        // data_size = 4 bytes
           | (1 << 20)        // pad_enable
           | (pad_code << 22) // pad_interval (6 -> 128 dw, 7 -> 256 dw)
           | (3 << 25);       // pad_amount = 4 dwords
    i32x8 g1;
    g1[0] = w0;
    g1[1] = (int)0xFFFF0000;                       // tensor_dim0 = 0x7FFFFFFF (lo16)
    g1[2] = (int)0xFFFF7FFF;                       // tensor_dim0 hi / tensor_dim1 lo
    g1[3] = (int)((tile_cols << 16) | 0x7FFF);     // tensor_dim1 hi | tile_dim0
    g1[4] = tile_rows;                             // tile_dim1 (tile_dim2 = 0)
    g1[5] = stride_elems;                          // tensor_dim0_stride[31:0]
    g1[6] = 0; g1[7] = 0;                          // stride hi / tensor_dim1_stride
    i32x4 gz4 = {0, 0, 0, 0};
    i32x8 gz8 = {0, 0, 0, 0, 0, 0, 0, 0};
    __builtin_amdgcn_tensor_load_to_lds(g0, g1, gz4, gz4, gz8, 0);
}
#endif

// ---------------- small prep kernels ----------------
__global__ void k_cast16(const float* __restrict__ src, _Float16* __restrict__ dst, int n) {
    int t = blockIdx.x * blockDim.x + threadIdx.x;
    if (t < n) dst[t] = (_Float16)src[t];
}

__global__ void k_packw1(const float* __restrict__ W1, _Float16* __restrict__ Wh) {
    int t = blockIdx.x * blockDim.x + threadIdx.x;   // over [H,F,D]
    if (t >= HEADS * F_IN * D_HEAD) return;
    int h = t / (F_IN * D_HEAD);
    int f = (t / D_HEAD) % F_IN;
    int d = t % D_HEAD;
    Wh[(size_t)f * F_HID + h * D_HEAD + d] = (_Float16)W1[t];
}

__global__ void k_elu16(const float* __restrict__ src, _Float16* __restrict__ dst, int n) {
    int t = blockIdx.x * blockDim.x + threadIdx.x;
    if (t < n) {
        float x = src[t];
        float y = x > 0.f ? x : (__expf(x) - 1.f);
        dst[t] = (_Float16)y;
    }
}

__global__ void k_esed(const float* __restrict__ Hm, const float* __restrict__ asrc,
                       const float* __restrict__ adst, float* __restrict__ es,
                       float* __restrict__ ed, int heads, int d) {
    int t = blockIdx.x * blockDim.x + threadIdx.x;
    if (t >= heads * N_NODES) return;
    int h = t / N_NODES, n = t % N_NODES;
    int ncol = heads * d;
    const float* hp = Hm + (size_t)n * ncol + h * d;
    float s = 0.f, q = 0.f;
    for (int k = 0; k < d; ++k) { s += hp[k] * asrc[h * d + k]; q += hp[k] * adst[h * d + k]; }
    es[t] = s; ed[t] = q;
}

// per-head max over e_dst, then stable-shift m[h,i] = leaky(es + maxdst) (upper bound)
__global__ void k_maxm(const float* __restrict__ es, const float* __restrict__ ed,
                       float* __restrict__ mrow) {
    __shared__ float red[256];
    int h = blockIdx.x;
    float m = -3.0e38f;
    for (int n = threadIdx.x; n < N_NODES; n += 256) m = fmaxf(m, ed[h * N_NODES + n]);
    red[threadIdx.x] = m; __syncthreads();
    for (int s = 128; s > 0; s >>= 1) {
        if (threadIdx.x < s) red[threadIdx.x] = fmaxf(red[threadIdx.x], red[threadIdx.x + s]);
        __syncthreads();
    }
    float md = red[0];
    for (int n = threadIdx.x; n < N_NODES; n += 256) {
        float t = es[h * N_NODES + n] + md;
        mrow[h * N_NODES + n] = t > 0.f ? t : LEAKY * t;
    }
}

// ---------------- GEMM1: H1[4096,64] = Xh[4096,512] @ W1h[512,64] ----------------
__global__ void __launch_bounds__(128) k_gemm1(const _Float16* __restrict__ Xh,
                                               const _Float16* __restrict__ Wh,
                                               float* __restrict__ H1,
                                               _Float16* __restrict__ H1h) {
    int wave = threadIdx.x >> 5;
    int lane = threadIdx.x & 31;
    int hi = lane >> 4, l16 = lane & 15;
    int row0 = blockIdx.x * 16;
    int col0 = wave * 16;
    int arow = row0 + l16;
    v8f acc = {};
    for (int kt = 0; kt < F_IN; kt += 32) {
        v16h a, b;
        const _Float16* ap = Xh + (size_t)arow * F_IN + kt + hi * 8;
        #pragma unroll
        for (int i = 0; i < 8; ++i) { a[i] = ap[i]; a[8 + i] = ap[16 + i]; }
        const _Float16* bp = Wh + (size_t)(kt + hi * 8) * F_HID + col0 + l16;
        #pragma unroll
        for (int i = 0; i < 8; ++i) { b[i] = bp[i * F_HID]; b[8 + i] = bp[(16 + i) * F_HID]; }
        acc = __builtin_amdgcn_wmma_f32_16x16x32_f16(false, a, false, b, (short)0, acc, false, false);
    }
    #pragma unroll
    for (int r = 0; r < 8; ++r) {
        int m = row0 + r + hi * 8;
        int n = col0 + l16;
        H1 [(size_t)m * F_HID + n] = acc[r];
        H1h[(size_t)m * F_HID + n] = (_Float16)acc[r];
    }
}

// ---------------- GEMM2: H2[4096,16] = Eh[4096,64] @ W2h[64,16] ----------------
__global__ void __launch_bounds__(32) k_gemm2(const _Float16* __restrict__ Eh,
                                              const _Float16* __restrict__ Wh,
                                              float* __restrict__ H2,
                                              _Float16* __restrict__ H2h) {
    int lane = threadIdx.x & 31;
    int hi = lane >> 4, l16 = lane & 15;
    int row0 = blockIdx.x * 16;
    int arow = row0 + l16;
    v8f acc = {};
    for (int kt = 0; kt < F_HID; kt += 32) {
        v16h a, b;
        const _Float16* ap = Eh + (size_t)arow * F_HID + kt + hi * 8;
        #pragma unroll
        for (int i = 0; i < 8; ++i) { a[i] = ap[i]; a[8 + i] = ap[16 + i]; }
        const _Float16* bp = Wh + (size_t)(kt + hi * 8) * NCLASS + l16;
        #pragma unroll
        for (int i = 0; i < 8; ++i) { b[i] = bp[i * NCLASS]; b[8 + i] = bp[(16 + i) * NCLASS]; }
        acc = __builtin_amdgcn_wmma_f32_16x16x32_f16(false, a, false, b, (short)0, acc, false, false);
    }
    #pragma unroll
    for (int r = 0; r < 8; ++r) {
        int m = row0 + r + hi * 8;
        H2 [(size_t)m * NCLASS + l16] = acc[r];
        H2h[(size_t)m * NCLASS + l16] = (_Float16)acc[r];
    }
}

// ---------------- Fused attention, layer 1 (8 heads, D=8) ----------------
// block = 16 rows; 8 waves, one per head. adj tiles [16x128] staged in LDS by
// the Tensor Data Mover (double buffered, padded rows -> stride 132 dwords),
// shared by all 8 waves. Single streaming pass, exact shifted softmax.
#define TJ1 128
#define RS1 132
__global__ void __launch_bounds__(256) k_attn1(const float* __restrict__ adj,
                                               const _Float16* __restrict__ H1h,
                                               const float* __restrict__ es,
                                               const float* __restrict__ ed,
                                               const float* __restrict__ mrow,
                                               float* __restrict__ out1) {
    __shared__ float atile[2][16 * RS1];
    int h = threadIdx.x >> 5;          // head = wave id
    int lane = threadIdx.x & 31;
    int hi = lane >> 4, l16 = lane & 15;
    int i0 = blockIdx.x * 16;
    int arow = i0 + l16;
    float esr = es  [h * N_NODES + arow];
    float mr  = mrow[h * N_NODES + arow];
    const float* gbase = adj + (size_t)i0 * N_NODES;
    const float* edh   = ed + h * N_NODES;
    int cc = l16 & 7;                  // duplicate B cols 8..15 (never stored)
    v8f acc = {};
    float psum = 0.f;
    const int NT = N_NODES / TJ1;
#if HAS_TDM
    if (h == 0) tdm_load_tile(gbase, &atile[0][0], TJ1, 16, N_NODES, 6);
#endif
    for (int t = 0; t < NT; ++t) {
        int buf;
#if HAS_TDM
        buf = t & 1;
        if (h == 0) {
            if (t + 1 < NT) {
                tdm_load_tile(gbase + (size_t)(t + 1) * TJ1, &atile[buf ^ 1][0], TJ1, 16, N_NODES, 6);
                __builtin_amdgcn_s_wait_tensorcnt(1);
            } else {
                __builtin_amdgcn_s_wait_tensorcnt(0);
            }
        }
        __syncthreads();
#else
        buf = 0;
        __syncthreads();
        for (int idx = threadIdx.x; idx < 16 * TJ1; idx += 256) {
            int r = idx / TJ1, c = idx % TJ1;
            atile[0][r * RS1 + c] = gbase[(size_t)r * N_NODES + t * TJ1 + c];
        }
        __syncthreads();
#endif
        const float* at = &atile[buf][0] + l16 * RS1;
        for (int jo = 0; jo < TJ1; jo += 32) {
            int jt = t * TJ1 + jo;
            v16h a, b;
            const float* adp = at + jo + hi * 8;
            const float* edp = edh + jt + hi * 8;
            #pragma unroll
            for (int i = 0; i < 8; ++i) {
                float t0 = esr + edp[i];
                float e0 = t0 > 0.f ? t0 : LEAKY * t0;
                float g0 = adp[i] > 0.f ? 1.f : 0.f;
                float p0 = g0 * __expf(e0 - mr);
                float t1 = esr + edp[16 + i];
                float e1 = t1 > 0.f ? t1 : LEAKY * t1;
                float g1 = adp[16 + i] > 0.f ? 1.f : 0.f;
                float p1 = g1 * __expf(e1 - mr);
                a[i] = (_Float16)p0; a[8 + i] = (_Float16)p1;
                psum += p0 + p1;
            }
            const _Float16* bp = H1h + (size_t)(jt + hi * 8) * F_HID + h * D_HEAD;
            #pragma unroll
            for (int i = 0; i < 8; ++i) { b[i] = bp[i * F_HID + cc]; b[8 + i] = bp[(16 + i) * F_HID + cc]; }
            acc = __builtin_amdgcn_wmma_f32_16x16x32_f16(false, a, false, b, (short)0, acc, false, false);
        }
#if HAS_TDM
        __syncthreads();   // all waves done with buf before TDM reuses it
#endif
    }
    #pragma unroll
    for (int r = 0; r < 8; ++r) {
        int m = r + hi * 8;
        float rs = __shfl(psum, m, 32) + __shfl(psum, m + 16, 32);
        if (l16 < 8)
            out1[(size_t)(i0 + m) * F_HID + h * D_HEAD + l16] = acc[r] / rs;
    }
}

// ---------------- Fused attention, layer 2 (1 head, D=16) ----------------
// block = 16 rows; one TDM-staged [16x256] adj tile shared by 8 waves, each
// wave consumes a 32-column strip; LDS reduce of partial tiles + row sums.
#define TJ2 256
#define RS2 260
__global__ void __launch_bounds__(256) k_attn2(const float* __restrict__ adj,
                                               const _Float16* __restrict__ H2h,
                                               const float* __restrict__ es,
                                               const float* __restrict__ ed,
                                               const float* __restrict__ mrow,
                                               float* __restrict__ out2) {
    __shared__ float atile[2][16 * RS2];
    __shared__ float accbuf[8][16][16];
    __shared__ float psbuf[8][32];
    int wave = threadIdx.x >> 5;
    int lane = threadIdx.x & 31;
    int hi = lane >> 4, l16 = lane & 15;
    int i0 = blockIdx.x * 16;
    int arow = i0 + l16;
    float esr = es[arow];
    float mr  = mrow[arow];
    const float* gbase = adj + (size_t)i0 * N_NODES;
    int c0 = wave * 32;                // this wave's column strip within a tile
    v8f acc = {};
    float psum = 0.f;
    const int NT = N_NODES / TJ2;
#if HAS_TDM
    if (wave == 0) tdm_load_tile(gbase, &atile[0][0], TJ2, 16, N_NODES, 7);
#endif
    for (int t = 0; t < NT; ++t) {
        int buf;
#if HAS_TDM
        buf = t & 1;
        if (wave == 0) {
            if (t + 1 < NT) {
                tdm_load_tile(gbase + (size_t)(t + 1) * TJ2, &atile[buf ^ 1][0], TJ2, 16, N_NODES, 7);
                __builtin_amdgcn_s_wait_tensorcnt(1);
            } else {
                __builtin_amdgcn_s_wait_tensorcnt(0);
            }
        }
        __syncthreads();
#else
        buf = 0;
        __syncthreads();
        for (int idx = threadIdx.x; idx < 16 * TJ2; idx += 256) {
            int r = idx / TJ2, c = idx % TJ2;
            atile[0][r * RS2 + c] = gbase[(size_t)r * N_NODES + t * TJ2 + c];
        }
        __syncthreads();
#endif
        {
            int jt = t * TJ2 + c0;
            v16h a, b;
            const float* adp = &atile[buf][0] + l16 * RS2 + c0 + hi * 8;
            const float* edp = ed + jt + hi * 8;
            #pragma unroll
            for (int i = 0; i < 8; ++i) {
                float t0 = esr + edp[i];
                float e0 = t0 > 0.f ? t0 : LEAKY * t0;
                float g0 = adp[i] > 0.f ? 1.f : 0.f;
                float p0 = g0 * __expf(e0 - mr);
                float t1 = esr + edp[16 + i];
                float e1 = t1 > 0.f ? t1 : LEAKY * t1;
                float g1 = adp[16 + i] > 0.f ? 1.f : 0.f;
                float p1 = g1 * __expf(e1 - mr);
                a[i] = (_Float16)p0; a[8 + i] = (_Float16)p1;
                psum += p0 + p1;
            }
            const _Float16* bp = H2h + (size_t)(jt + hi * 8) * NCLASS + l16;
            #pragma unroll
            for (int i = 0; i < 8; ++i) { b[i] = bp[i * NCLASS]; b[8 + i] = bp[(16 + i) * NCLASS]; }
            acc = __builtin_amdgcn_wmma_f32_16x16x32_f16(false, a, false, b, (short)0, acc, false, false);
        }
        __syncthreads();   // all waves done with buf before TDM reuses it
    }
    #pragma unroll
    for (int r = 0; r < 8; ++r) accbuf[wave][r + hi * 8][l16] = acc[r];
    psbuf[wave][lane] = psum;
    __syncthreads();
    int t = threadIdx.x;
    int m = t >> 4, n = t & 15;
    float s = 0.f, rs = 0.f;
    #pragma unroll
    for (int w = 0; w < 8; ++w) { s += accbuf[w][m][n]; rs += psbuf[w][m] + psbuf[w][m + 16]; }
    out2[(size_t)(i0 + m) * NCLASS + n] = s / rs;
}

// ---------------- in-place log-softmax over 16 classes ----------------
__global__ void k_logsm(float* __restrict__ out) {
    int n = blockIdx.x * blockDim.x + threadIdx.x;
    if (n >= N_NODES) return;
    float v[NCLASS]; float m = -3.0e38f;
    #pragma unroll
    for (int c = 0; c < NCLASS; ++c) { v[c] = out[(size_t)n * NCLASS + c]; m = fmaxf(m, v[c]); }
    float s = 0.f;
    #pragma unroll
    for (int c = 0; c < NCLASS; ++c) s += __expf(v[c] - m);
    float ls = __logf(s);
    #pragma unroll
    for (int c = 0; c < NCLASS; ++c) out[(size_t)n * NCLASS + c] = v[c] - m - ls;
}

extern "C" void kernel_launch(void* const* d_in, const int* in_sizes, int n_in,
                              void* d_out, int out_size, void* d_ws, size_t ws_size,
                              hipStream_t stream) {
    const float* x   = (const float*)d_in[0];
    const float* adj = (const float*)d_in[1];
    const float* W1  = (const float*)d_in[2];
    const float* a1s = (const float*)d_in[3];
    const float* a1d = (const float*)d_in[4];
    const float* W2  = (const float*)d_in[5];
    const float* a2s = (const float*)d_in[6];
    const float* a2d = (const float*)d_in[7];
    float* out = (float*)d_out;

    char* w = (char*)d_ws;
    _Float16* Xh   = (_Float16*)(w + OFF_XH);
    _Float16* W1h  = (_Float16*)(w + OFF_W1H);
    float*    H1   = (float*)   (w + OFF_H1);
    _Float16* H1h  = (_Float16*)(w + OFF_H1H);
    float*    es1  = (float*)   (w + OFF_ES1);
    float*    ed1  = (float*)   (w + OFF_ED1);
    float*    mr1  = (float*)   (w + OFF_MR1);
    float*    out1 = (float*)   (w + OFF_OUT1);
    _Float16* Eh   = (_Float16*)(w + OFF_EH);
    _Float16* W2h  = (_Float16*)(w + OFF_W2H);
    float*    H2   = (float*)   (w + OFF_H2);
    _Float16* H2h  = (_Float16*)(w + OFF_H2H);
    float*    es2  = (float*)   (w + OFF_ES2);
    float*    ed2  = (float*)   (w + OFF_ED2);
    float*    mr2  = (float*)   (w + OFF_MR2);

    // ---- layer 1 ----
    k_cast16<<<(N_NODES * F_IN + 255) / 256, 256, 0, stream>>>(x, Xh, N_NODES * F_IN);
    k_packw1<<<(HEADS * F_IN * D_HEAD + 255) / 256, 256, 0, stream>>>(W1, W1h);
    k_cast16<<<(F_HID * NCLASS + 255) / 256, 256, 0, stream>>>(W2, W2h, F_HID * NCLASS);
    k_gemm1<<<N_NODES / 16, 128, 0, stream>>>(Xh, W1h, H1, H1h);
    k_esed<<<(HEADS * N_NODES + 255) / 256, 256, 0, stream>>>(H1, a1s, a1d, es1, ed1, HEADS, D_HEAD);
    k_maxm<<<HEADS, 256, 0, stream>>>(es1, ed1, mr1);
    k_attn1<<<N_NODES / 16, 256, 0, stream>>>(adj, H1h, es1, ed1, mr1, out1);

    // ---- layer 2 ----
    k_elu16<<<(N_NODES * F_HID + 255) / 256, 256, 0, stream>>>(out1, Eh, N_NODES * F_HID);
    k_gemm2<<<N_NODES / 16, 32, 0, stream>>>(Eh, W2h, H2, H2h);
    k_esed<<<(N_NODES + 255) / 256, 256, 0, stream>>>(H2, a2s, a2d, es2, ed2, 1, NCLASS);
    k_maxm<<<1, 256, 0, stream>>>(es2, ed2, mr2);
    k_attn2<<<N_NODES / 16, 256, 0, stream>>>(adj, H2h, es2, ed2, mr2, out);
    k_logsm<<<(N_NODES + 255) / 256, 256, 0, stream>>>(out);
}
